// EnhancedMemoryStack_24902220382488
// MI455X (gfx1250) — compile-verified, matrix operand order checked by reading, not run on hardware
//
#include <hip/hip_runtime.h>
#include <hip/hip_bf16.h>

typedef __attribute__((ext_vector_type(16))) _Float16 v16h;
typedef __attribute__((ext_vector_type(8)))  float    v8f;

// Problem shape
#define BSZ 32768
#define SLOTS 16
#define DIM 64
#define TDIM 128

// Launch shape: 1024 blocks x 4 waves x 8 iters = 32768 items
#define WAVES 4
#define NBLK 1024
#define ITERS 8

// d_out flat offsets (return order: z_read_r, z_read_i, mem_new, new_ptr, active_slots)
#define O_ZR  0LL
#define O_ZI  ((long long)BSZ * DIM)
#define O_MEM (2LL * BSZ * DIM)
#define O_PTR (O_MEM + (long long)BSZ * SLOTS * TDIM)
#define O_ACT (O_PTR + (long long)BSZ * SLOTS)

// Per-wave LDS scratch layout (bytes)
#define WFRAG_BYTES 49152              // 6 mats * 4 etiles * 2 ksteps * 512 halves * 2B
#define XF_OFF 0                        // 4 A-frags (mr ks0/1, mi ks0/1)      4096 B
#define QF_OFF 4096                     // 4 A-frags of q_flat (K=128)         4096 B
#define KF_OFF 8192                     // 4 B-frags of k_flat                 4096 B
#define VF_OFF 12288                    // 8 B-frags vr/vi (K padded to 32)    8192 B
#define AF_OFF 20480                    // 1 A-frag of attn (K padded to 32)   1024 B
#define PB_OFF 21504                    // 16 floats new_ptr                     64 B
#define WAVE_STRIDE 21568
#define SMEM_TOTAL (WFRAG_BYTES + WAVES * WAVE_STRIDE)   // 135424 B

// ISA 16-bit fragment index: lane = laneHi*16 + (m or n), h = (k>>4)*8 + (k&7),
// laneHi = (k>>3)&1.  Returns half-offset within a 512-half fragment.
__device__ __forceinline__ int frag_idx(int mn, int k) {
    int laneHi = (k >> 3) & 1;
    int grp    = k >> 4;
    int pos    = k & 7;
    return (laneHi * 16 + mn) * 16 + grp * 8 + pos;
}

__device__ __forceinline__ v8f wmma16(v16h a, v16h b, v8f c) {
    return __builtin_amdgcn_wmma_f32_16x16x32_f16(
        /*neg_a=*/false, a, /*neg_b=*/false, b,
        /*c_mod=*/(short)0, c, /*reuse_a=*/false, /*reuse_b=*/false);
}

__global__ void ems_zero_scalar(float* out) {
    out[O_ACT] = 0.0f;
}

__global__ __launch_bounds__(WAVES * 32)
void ems_main(const float* __restrict__ zr_in, const float* __restrict__ zi_in,
              const float* __restrict__ mem_in, const float* __restrict__ ptr_in,
              const float* __restrict__ ctrl_in,
              const float* __restrict__ w0, const float* __restrict__ w1,
              const float* __restrict__ w2, const float* __restrict__ w3,
              const float* __restrict__ w4, const float* __restrict__ w5,
              float* __restrict__ out) {
    extern __shared__ char smem[];
    _Float16* wfrag = (_Float16*)smem;

    const int tid  = threadIdx.x;
    const int lane = tid & 31;
    const int wid  = tid >> 5;

    char* wb = smem + WFRAG_BYTES + (size_t)wid * WAVE_STRIDE;
    _Float16* xf = (_Float16*)(wb + XF_OFF);
    _Float16* qf = (_Float16*)(wb + QF_OFF);
    _Float16* kf = (_Float16*)(wb + KF_OFF);
    _Float16* vf = (_Float16*)(wb + VF_OFF);
    _Float16* af = (_Float16*)(wb + AF_OFF);
    float*    pb = (float*)(wb + PB_OFF);

    // ---- Stage all 6 weight matrices into LDS as f16, pre-permuted into
    // B-fragment layout: B[k=d][n=e] = W[e][d] (y = X * W^T).
    const float* wmat[6] = {w0, w1, w2, w3, w4, w5};
    for (int idx = tid; idx < 6 * 4096; idx += blockDim.x) {
        int p   = idx >> 12;
        int rem = idx & 4095;
        int et  = rem >> 10;
        int ks  = (rem >> 9) & 1;
        int pos = rem & 511;
        int ln  = pos >> 4;
        int h   = pos & 15;
        int n      = ln & 15;
        int laneHi = ln >> 4;
        int kk  = (h & 7) + 8 * laneHi + 16 * (h >> 3);
        int e = et * 16 + n;
        int d = ks * 32 + kk;
        wfrag[idx] = (_Float16)wmat[p][e * 64 + d];
    }
    __syncthreads();

    const int laneHi = lane >> 4;
    const int nlo    = lane & 15;

#pragma unroll 1
    for (int it = 0; it < ITERS; ++it) {
        const int b = it * (NBLK * WAVES) + blockIdx.x * WAVES + wid;

        // ---- gates ----
        float c0 = ctrl_in[b * 3 + 0];
        float c1 = ctrl_in[b * 3 + 1];
        float c2 = ctrl_in[b * 3 + 2];
        float g0 = 1.0f / (1.0f + __expf(-c0));
        float g1 = 1.0f / (1.0f + __expf(-c1));
        float g2 = 1.0f / (1.0f + __expf(-c2));
        float tot  = g0 + g1 + g2 + 1e-6f;
        float push = g0 / tot, pop = g1 / tot, stay = g2 / tot;

        // ---- pointer update + active_slots ----
        float np = 0.0f;
        if (lane < 16) {
            float pc = ptr_in[b * 16 + lane];
            float pu = ptr_in[b * 16 + ((lane + 15) & 15)];   // roll +1
            float pd = ptr_in[b * 16 + ((lane + 1) & 15)];    // roll -1
            np = push * pu + pop * pd + stay * pc;
            out[O_PTR + (long long)b * 16 + lane] = np;
            pb[lane] = np;
        }
        float cnt = (lane < 16 && np > 0.1f) ? 1.0f : 0.0f;
        cnt += __shfl_xor(cnt, 1, 32);
        cnt += __shfl_xor(cnt, 2, 32);
        cnt += __shfl_xor(cnt, 4, 32);
        cnt += __shfl_xor(cnt, 8, 32);
        cnt += __shfl_xor(cnt, 16, 32);
        if (lane == 0) atomicAdd(out + O_ACT, cnt * (1.0f / (float)BSZ));

        // ---- zero padded fragment buffers (vf: K 16..31 zero; af: K 16..31 zero)
        {
            int* zv = (int*)vf;
#pragma unroll
            for (int i = 0; i < 64; ++i) zv[lane + i * 32] = 0;
            int* za = (int*)af;
#pragma unroll
            for (int i = 0; i < 8; ++i) za[lane + i * 32] = 0;
        }

        // ---- mem blend -> global mem_new (coalesced) + f16 A-fragments in LDS
        const float keep = 1.0f - push;
#pragma unroll 4
        for (int flat = lane; flat < SLOTS * TDIM; flat += 32) {
            int s = flat >> 7;
            int j = flat & 127;
            float mv = mem_in[(long long)b * 2048 + flat];
            float zv = (j < 64) ? zr_in[(long long)b * 64 + j]
                                : zi_in[(long long)b * 64 + (j - 64)];
            float mn = mv * keep + push * zv;
            out[O_MEM + (long long)b * 2048 + flat] = mn;
            int part = j >> 6;       // 0 = mr, 1 = mi
            int dd   = j & 63;
            int ks   = dd >> 5;
            xf[(part * 2 + ks) * 512 + frag_idx(s, dd & 31)] = (_Float16)mn;
        }
        if (it + 1 < ITERS)
            __builtin_prefetch(&mem_in[(long long)(b + NBLK * WAVES) * 2048 + lane * 64], 0, 1);

        // ---- load A fragments (contiguous 32B per lane) ----
        v16h amr[2], ami[2], ani[2];
        amr[0] = *(const v16h*)(xf + 0 * 512 + lane * 16);
        amr[1] = *(const v16h*)(xf + 1 * 512 + lane * 16);
        ami[0] = *(const v16h*)(xf + 2 * 512 + lane * 16);
        ami[1] = *(const v16h*)(xf + 3 * 512 + lane * 16);
#pragma unroll
        for (int ks = 0; ks < 2; ++ks)
#pragma unroll
            for (int i = 0; i < 16; ++i) ani[ks][i] = (_Float16)(-(float)ami[ks][i]);

        // ---- Q/K/V complex linears: 96 WMMAs ----
#pragma unroll
        for (int proj = 0; proj < 3; ++proj) {
#pragma unroll
            for (int et = 0; et < 4; ++et) {
                v8f accr = {};
                v8f acci = {};
#pragma unroll
                for (int ks = 0; ks < 2; ++ks) {
                    v16h bwr = *(const v16h*)(wfrag + ((2 * proj) * 8 + et * 2 + ks) * 512 + lane * 16);
                    v16h bwi = *(const v16h*)(wfrag + ((2 * proj + 1) * 8 + et * 2 + ks) * 512 + lane * 16);
                    accr = wmma16(amr[ks], bwr, accr);   // + mr * Wr^T
                    accr = wmma16(ani[ks], bwi, accr);   // - mi * Wi^T
                    acci = wmma16(amr[ks], bwi, acci);   // + mr * Wi^T
                    acci = wmma16(ami[ks], bwr, acci);   // + mi * Wr^T
                }
                const int m0 = 8 * laneHi;
                if (proj == 0) {          // q_flat -> A-fragment layout (rows=s, K=e 0..127)
#pragma unroll
                    for (int r = 0; r < 8; ++r) {
                        int m  = r + m0;
                        int er = et * 16 + nlo;
                        int ei = er + 64;
                        qf[(er >> 5) * 512 + frag_idx(m, er & 31)] = (_Float16)accr[r];
                        qf[(ei >> 5) * 512 + frag_idx(m, ei & 31)] = (_Float16)acci[r];
                    }
                } else if (proj == 1) {   // k_flat -> B-fragment layout (cols=t, K=e)
#pragma unroll
                    for (int r = 0; r < 8; ++r) {
                        int t  = r + m0;
                        int er = et * 16 + nlo;
                        int ei = er + 64;
                        kf[(er >> 5) * 512 + frag_idx(t, er & 31)] = (_Float16)accr[r];
                        kf[(ei >> 5) * 512 + frag_idx(t, ei & 31)] = (_Float16)acci[r];
                    }
                } else {                  // v -> B-fragment layout (cols=d, K=t, padded to 32)
#pragma unroll
                    for (int r = 0; r < 8; ++r) {
                        int t = r + m0;
                        vf[et * 512       + frag_idx(nlo, t)] = (_Float16)accr[r];
                        vf[(4 + et) * 512 + frag_idx(nlo, t)] = (_Float16)acci[r];
                    }
                }
            }
        }

        // ---- scores = q_flat * k_flat^T (16x16, K=128): 4 WMMAs ----
        v8f accs = {};
#pragma unroll
        for (int f = 0; f < 4; ++f) {
            v16h aq = *(const v16h*)(qf + f * 512 + lane * 16);
            v16h bk = *(const v16h*)(kf + f * 512 + lane * 16);
            accs = wmma16(aq, bk, accs);
        }

        // ---- softmax over t (cross-lane within 16-lane half), scatter to A-frag
        const float scale = 0.125f;   // D^-0.5
#pragma unroll
        for (int r = 0; r < 8; ++r) {
            float v = accs[r] * scale;
            float mx = v;
            mx = fmaxf(mx, __shfl_xor(mx, 1, 32));
            mx = fmaxf(mx, __shfl_xor(mx, 2, 32));
            mx = fmaxf(mx, __shfl_xor(mx, 4, 32));
            mx = fmaxf(mx, __shfl_xor(mx, 8, 32));
            float e  = __expf(v - mx);
            float sm = e;
            sm += __shfl_xor(sm, 1, 32);
            sm += __shfl_xor(sm, 2, 32);
            sm += __shfl_xor(sm, 4, 32);
            sm += __shfl_xor(sm, 8, 32);
            af[frag_idx(r + 8 * laneHi, nlo)] = (_Float16)(e / sm);
        }

        // ---- attn * V (8 WMMAs) + ptr-weighted read ----
        v16h a_at = *(const v16h*)(af + lane * 16);
#pragma unroll
        for (int et = 0; et < 4; ++et) {
            v16h bvr = *(const v16h*)(vf + et * 512 + lane * 16);
            v16h bvi = *(const v16h*)(vf + (4 + et) * 512 + lane * 16);
            v8f z8 = {};
            v8f ar = wmma16(a_at, bvr, z8);
            v8f ai = wmma16(a_at, bvi, z8);
            float pr = 0.0f, pi = 0.0f;
#pragma unroll
            for (int r = 0; r < 8; ++r) {
                float w = pb[r + 8 * laneHi];
                pr += w * ar[r];
                pi += w * ai[r];
            }
            pr += __shfl_xor(pr, 16, 32);
            pi += __shfl_xor(pi, 16, 32);
            if (lane < 16) {
                out[O_ZR + (long long)b * 64 + et * 16 + lane] = pr;
                out[O_ZI + (long long)b * 64 + et * 16 + lane] = pi;
            }
        }
    }
}

extern "C" void kernel_launch(void* const* d_in, const int* in_sizes, int n_in,
                              void* d_out, int out_size, void* d_ws, size_t ws_size,
                              hipStream_t stream) {
    (void)in_sizes; (void)n_in; (void)out_size; (void)d_ws; (void)ws_size;
    const float* zr   = (const float*)d_in[0];
    const float* zi   = (const float*)d_in[1];
    const float* mem  = (const float*)d_in[2];
    const float* ptr  = (const float*)d_in[3];
    const float* ctrl = (const float*)d_in[4];
    const float* w0 = (const float*)d_in[5];   // wq_r
    const float* w1 = (const float*)d_in[6];   // wq_i
    const float* w2 = (const float*)d_in[7];   // wk_r
    const float* w3 = (const float*)d_in[8];   // wk_i
    const float* w4 = (const float*)d_in[9];   // wv_r
    const float* w5 = (const float*)d_in[10];  // wv_i
    float* out = (float*)d_out;

    ems_zero_scalar<<<1, 1, 0, stream>>>(out);
    ems_main<<<dim3(NBLK), dim3(WAVES * 32), SMEM_TOTAL, stream>>>(
        zr, zi, mem, ptr, ctrl, w0, w1, w2, w3, w4, w5, out);
}
